// GTG_Module_13554916786313
// MI455X (gfx1250) — matrix operand order repaired and checked by reading, not compiled, not run on hardware
//
#include <hip/hip_runtime.h>

#define NROW 8192
#define EDIM 512
#define NCLS 10
#define MAXIT 30
#define TOLV 0.001f
#define BNEPS 1e-5f
#define LOGEPS 1e-20f

typedef __attribute__((ext_vector_type(16))) __bf16 v16bf;
typedef __attribute__((ext_vector_type(8)))  __bf16 v8bf;
typedef __attribute__((ext_vector_type(8)))  float  v8f;

// ---- WMMA operand loaders (CDNA5 16-bit A/B layouts, wave32) ----
// A 16x32: lane holds row (lane&15); K halves {c..c+7, c+16..c+23}, c=(lane>>4)*8
static __device__ inline v16bf load_a_op(const __bf16* rowptr, int k, int lane) {
    const __bf16* p = rowptr + k + ((lane >> 4) << 3);
    union { v16bf v; v8bf h[2]; } u;
    u.h[0] = *(const v8bf*)(p);
    u.h[1] = *(const v8bf*)(p + 16);
    return u.v;
}
// B 32x16: lane holds col (lane&15); K halves {c..c+7, c+16..c+23}, c=(lane>>4)*8
static __device__ inline v16bf load_b_op(const __bf16* rowptr, int k, int lane) {
    const __bf16* p = rowptr + k + ((lane >> 4) << 3);
    union { v16bf v; v8bf h[2]; } u;
    u.h[0] = *(const v8bf*)(p);
    u.h[1] = *(const v8bf*)(p + 16);
    return u.v;
}

// LDS variants (row stride 72 halfs = 144B -> bank stride 36, conflict-free over 16 lanes)
static __device__ inline v16bf lds_a_op(const __bf16 (*s)[72], int row0, int koff, int lane) {
    const __bf16* p = &s[row0 + (lane & 15)][koff + ((lane >> 4) << 3)];
    union { v16bf v; v8bf h[2]; } u;
    u.h[0] = *(const v8bf*)(p);
    u.h[1] = *(const v8bf*)(p + 16);
    return u.v;
}
static __device__ inline v16bf lds_b_op(const __bf16 (*s)[72], int n0, int koff, int lane) {
    const __bf16* p = &s[n0 + (lane & 15)][koff + ((lane >> 4) << 3)];
    union { v16bf v; v8bf h[2]; } u;
    u.h[0] = *(const v8bf*)(p);
    u.h[1] = *(const v8bf*)(p + 16);
    return u.v;
}

// CDNA5 async global->LDS copy, 16B per lane (tracked by ASYNCcnt)
static __device__ inline void async_cp16(const __bf16* g, const __bf16* l) {
    const unsigned ldsaddr = (unsigned)(uintptr_t)l;   // low 32 bits of generic = LDS offset
    asm volatile("global_load_async_to_lds_b128 %0, %1, off"
                 :: "v"(ldsaddr), "v"(g) : "memory");
}

static __device__ inline void ins7(float (&s)[7], float v) {
    if (v < s[6]) {
        s[6] = v;
        for (int q = 6; q > 0; --q) {
            if (s[q] < s[q - 1]) { float t = s[q - 1]; s[q - 1] = s[q]; s[q] = t; }
            else break;
        }
    }
}

// ---------------- elementwise helpers ----------------
__global__ void k_f32_to_bf16(const float* __restrict__ src, __bf16* __restrict__ dst, int n) {
    int i = blockIdx.x * blockDim.x + threadIdx.x;
    if (i < n) dst[i] = (__bf16)src[i];
}

__global__ void k_rownorm(const float* __restrict__ emb, float* __restrict__ sq) {
    const int lane = threadIdx.x & 31;
    const int row  = blockIdx.x * (blockDim.x >> 5) + (threadIdx.x >> 5);
    if (row >= NROW) return;
    const float* r = emb + (size_t)row * EDIM;
    float acc = 0.f;
    for (int k = lane; k < EDIM; k += 32) { float v = r[k]; acc += v * v; }
    for (int off = 16; off; off >>= 1) acc += __shfl_xor(acc, off, 32);
    if (lane == 0) sq[row] = acc;
}

// ---------------- pass A: sigma_i = dist to 7th NN (incl self) ----------------
// Two column tiles per wave => two independent load->wmma chains (overlap).
__global__ void __launch_bounds__(128) k_sigma(const __bf16* __restrict__ embh,
                                               const float* __restrict__ sq,
                                               float* __restrict__ sigma) {
    __shared__ float tile[4][16][17];
    __shared__ float cand[16][56];
    const int lane = threadIdx.x & 31;
    const int wave = threadIdx.x >> 5;
    const int rowBase = blockIdx.x * 16;
    const __bf16* arow = embh + (size_t)(rowBase + (lane & 15)) * EDIM;
    const int myrow = lane >> 1;
    const int hsel  = lane & 1;
    float top[7];
#pragma unroll
    for (int q = 0; q < 7; ++q) top[q] = 3.0e38f;

    for (int t = wave * 2; t < NROW / 16; t += 8) {       // uniform trip count: 64 pairs
        const int cb0 = t * 16;
        const int cb1 = (t + 1) * 16;
        const __bf16* brow0 = embh + (size_t)(cb0 + (lane & 15)) * EDIM;
        const __bf16* brow1 = embh + (size_t)(cb1 + (lane & 15)) * EDIM;
        v8f acc0 = {}, acc1 = {};
        for (int k = 0; k < EDIM; k += 32) {
            v16bf a  = load_a_op(arow, k, lane);
            v16bf b0 = load_b_op(brow0, k, lane);
            v16bf b1 = load_b_op(brow1, k, lane);
            acc0 = __builtin_amdgcn_wmma_f32_16x16x32_bf16(false, a, false, b0,
                                                           (short)0, acc0, false, false);
            acc1 = __builtin_amdgcn_wmma_f32_16x16x32_bf16(false, a, false, b1,
                                                           (short)0, acc1, false, false);
        }
        const int mo = (lane >> 4) << 3;
#pragma unroll
        for (int pair = 0; pair < 2; ++pair) {
            const v8f acc = pair ? acc1 : acc0;
            const int n   = (pair ? cb1 : cb0) + (lane & 15);
            const float sqn = sq[n];
#pragma unroll
            for (int r = 0; r < 8; ++r) {
                const int m = rowBase + r + mo;
                float d2 = sq[m] + sqn - 2.f * acc[r];
                d2 = d2 > 0.f ? d2 : 0.f;
                tile[wave][r + mo][lane & 15] = d2;
            }
            __syncthreads();
#pragma unroll
            for (int c = 0; c < 8; ++c) ins7(top, tile[wave][myrow][hsel * 8 + c]);
            __syncthreads();
        }
    }
    const int slot = wave * 2 + hsel;
#pragma unroll
    for (int q = 0; q < 7; ++q) cand[myrow][slot * 7 + q] = top[q];
    __syncthreads();
    if (threadIdx.x < 16) {
        float s[7];
#pragma unroll
        for (int q = 0; q < 7; ++q) s[q] = 3.0e38f;
        for (int j = 0; j < 56; ++j) ins7(s, cand[threadIdx.x][j]);
        sigma[rowBase + threadIdx.x] = sqrtf(s[6]);
    }
}

// ---------------- pass B: A = exp(-D^2/(sig_i sig_j)) as bf16 ----------------
// 64x64 macro-tile; async double-buffered LDS staging of 64x64-half k-slices
// (two WMMA k-steps per stage -> 8 WMMAs between barriers).
__global__ void __launch_bounds__(128) k_gram(const __bf16* __restrict__ embh,
                                              const float* __restrict__ sq,
                                              const float* __restrict__ sigma,
                                              __bf16* __restrict__ A) {
    __shared__ __bf16 sA[2][64][72];
    __shared__ __bf16 sB[2][64][72];
    const int tid  = threadIdx.x;
    const int lane = tid & 31;
    const int wave = tid >> 5;
    const int rowBase = blockIdx.y * 64;
    const int colBase = blockIdx.x * 64;

    // stage: 64 rows x 64 halfs = 512 chunks of 8 halfs per matrix; 4 per thread
    auto stage = [&](int k, int buf) {
#pragma unroll
        for (int c = 0; c < 4; ++c) {
            const int chunk = tid + c * 128;
            const int row = chunk >> 3;
            const int kc  = (chunk & 7) << 3;
            async_cp16(embh + (size_t)(rowBase + row) * EDIM + k + kc, &sA[buf][row][kc]);
            async_cp16(embh + (size_t)(colBase + row) * EDIM + k + kc, &sB[buf][row][kc]);
        }
    };

    v8f accs[4] = {};
    stage(0, 0);
    const int NK = EDIM / 64;                 // 8 stages
    for (int ks = 0; ks < NK; ++ks) {
        const int cur = ks & 1;
        if (ks + 1 < NK) {
            stage((ks + 1) * 64, cur ^ 1);
            asm volatile("s_wait_asynccnt 0x8" ::: "memory");   // stage 'cur' landed
        } else {
            asm volatile("s_wait_asynccnt 0x0" ::: "memory");
        }
        __syncthreads();
#pragma unroll
        for (int sub = 0; sub < 2; ++sub) {
            v16bf a = lds_a_op(sA[cur], wave * 16, sub * 32, lane);
#pragma unroll
            for (int nt = 0; nt < 4; ++nt) {
                v16bf b = lds_b_op(sB[cur], nt * 16, sub * 32, lane);
                accs[nt] = __builtin_amdgcn_wmma_f32_16x16x32_bf16(false, a, false, b,
                                                                   (short)0, accs[nt],
                                                                   false, false);
            }
        }
        __syncthreads();
    }

    const int mo = (lane >> 4) << 3;
    const int rb = rowBase + wave * 16;
#pragma unroll
    for (int nt = 0; nt < 4; ++nt) {
        const int n = colBase + nt * 16 + (lane & 15);
        const float sqn = sq[n];
        const float sgn = sigma[n];
#pragma unroll
        for (int r = 0; r < 8; ++r) {
            const int m = rb + r + mo;
            float d2 = sq[m] + sqn - 2.f * accs[nt][r];
            d2 = d2 > 0.f ? d2 : 0.f;
            float av = __expf(-d2 / (sigma[m] * sgn));
            A[(size_t)m * NROW + n] = (__bf16)av;
        }
    }
}

// ---------------- replicator dynamics ----------------
__global__ void k_init(const int* __restrict__ labels, const int* __restrict__ mask,
                       float* __restrict__ X0, float* __restrict__ entsum,
                       int* __restrict__ done, float* __restrict__ errsq) {
    int i = blockIdx.x * blockDim.x + threadIdx.x;
    if (i == 0) { *done = 0; *errsq = 0.f; }
    if (i < NROW) {
        entsum[i] = 0.f;
        const int lab = mask[i];
        const int l = labels[i];
#pragma unroll
        for (int c = 0; c < NCLS; ++c)
            X0[i * NCLS + c] = lab ? (c == l ? 1.f : 0.f) : (1.f / NCLS);
    }
}

__global__ void __launch_bounds__(256) k_replicator(const __bf16* __restrict__ A,
                                                    const float* __restrict__ Xold,
                                                    float* __restrict__ Xnew,
                                                    float* __restrict__ entsum,
                                                    const int* __restrict__ done,
                                                    float* __restrict__ errsq) {
    const int lane = threadIdx.x & 31;
    const int row  = blockIdx.x * 8 + (threadIdx.x >> 5);
    if (*done) {                                   // masked iteration: X passes through
        if (lane < NCLS) Xnew[(size_t)row * NCLS + lane] = Xold[(size_t)row * NCLS + lane];
        return;
    }
    const __bf16* arow = A + (size_t)row * NROW;
    float acc[NCLS];
#pragma unroll
    for (int c = 0; c < NCLS; ++c) acc[c] = 0.f;
    for (int jb = 0; jb < NROW; jb += 256) {       // 16B of A per lane per step
        const int j0 = jb + lane * 8;
        const v8bf av = *(const v8bf*)(arow + j0);
        __builtin_prefetch(arow + j0 + 2048, 0, 1);
#pragma unroll
        for (int u = 0; u < 8; ++u) {
            const float a = (float)av[u];
            const float* xr = Xold + (size_t)(j0 + u) * NCLS;
#pragma unroll
            for (int c = 0; c < NCLS; ++c) acc[c] = fmaf(a, xr[c], acc[c]);
        }
    }
#pragma unroll
    for (int c = 0; c < NCLS; ++c)
        for (int off = 16; off; off >>= 1) acc[c] += __shfl_xor(acc[c], off, 32);
    if (lane == 0) {
        const float* xo = Xold + (size_t)row * NCLS;
        float xn[NCLS]; float s = 0.f;
#pragma unroll
        for (int c = 0; c < NCLS; ++c) { xn[c] = xo[c] * acc[c]; s += xn[c]; }
        const float inv = 1.f / s;
        float d2 = 0.f, ent = 0.f;
#pragma unroll
        for (int c = 0; c < NCLS; ++c) {
            float v = xn[c] * inv;
            float df = v - xo[c];
            d2 += df * df;
            ent -= v * __logf(v + LOGEPS);
            Xnew[(size_t)row * NCLS + c] = v;
        }
        entsum[row] += ent;
        atomicAdd(errsq, d2);
    }
}

__global__ void k_update(int* __restrict__ done, float* __restrict__ errsq) {
    if (sqrtf(*errsq) <= TOLV) *done = 1;
    *errsq = 0.f;
}

// ---------------- MLP ----------------
__global__ void __launch_bounds__(128) k_gemm_bias(const __bf16* __restrict__ Xb,
                                                   const __bf16* __restrict__ Wb,
                                                   const float* __restrict__ bias,
                                                   float* __restrict__ Y, int K, int O) {
    const int lane = threadIdx.x & 31;
    const int wave = threadIdx.x >> 5;
    const int rowBase = blockIdx.y * 64 + wave * 16;
    const int colBase = blockIdx.x * 16;
    const __bf16* arow = Xb + (size_t)(rowBase + (lane & 15)) * K;
    const __bf16* brow = Wb + (size_t)(colBase + (lane & 15)) * K;
    v8f acc = {};
    for (int k = 0; k < K; k += 32) {
        v16bf a = load_a_op(arow, k, lane);
        v16bf b = load_b_op(brow, k, lane);
        acc = __builtin_amdgcn_wmma_f32_16x16x32_bf16(false, a, false, b,
                                                      (short)0, acc, false, false);
    }
    const int mo = (lane >> 4) << 3;
    const int n  = colBase + (lane & 15);
    const float bv = bias[n];
#pragma unroll
    for (int r = 0; r < 8; ++r)
        Y[(size_t)(rowBase + r + mo) * O + n] = acc[r] + bv;
}

__global__ void k_bn_stats(const float* __restrict__ Y, float* __restrict__ stats, int O) {
    __shared__ float s1[256], s2[256];
    const int o = blockIdx.x;
    float a = 0.f, b = 0.f;
    for (int r = threadIdx.x; r < NROW; r += 256) {
        float v = Y[(size_t)r * O + o];
        a += v; b += v * v;
    }
    s1[threadIdx.x] = a; s2[threadIdx.x] = b;
    __syncthreads();
    for (int off = 128; off; off >>= 1) {
        if (threadIdx.x < off) { s1[threadIdx.x] += s1[threadIdx.x + off];
                                 s2[threadIdx.x] += s2[threadIdx.x + off]; }
        __syncthreads();
    }
    if (threadIdx.x == 0) {
        float mu = s1[0] / NROW;
        stats[o] = mu;
        stats[O + o] = s2[0] / NROW - mu * mu;
    }
}

__global__ void k_bn_apply(const float* __restrict__ Y, const float* __restrict__ stats,
                           const float* __restrict__ g, const float* __restrict__ beta,
                           __bf16* __restrict__ outb, float* __restrict__ outf, int O) {
    int i = blockIdx.x * blockDim.x + threadIdx.x;
    if (i >= NROW * O) return;
    int o = i % O;
    float v = (Y[i] - stats[o]) * rsqrtf(stats[O + o] + BNEPS) * g[o] + beta[o];
    v = v > 0.f ? v : 0.f;
    outb[i] = (__bf16)v;
    outf[i] = v;
}

__global__ void k_final(const float* __restrict__ act, const float* __restrict__ Wf,
                        const float* __restrict__ bf, float* __restrict__ ypred) {
    int i = blockIdx.x * blockDim.x + threadIdx.x;
    if (i >= NROW) return;
    float s = bf[0];
#pragma unroll
    for (int k = 0; k < 16; ++k) s = fmaf(act[i * 16 + k], Wf[k], s);
    ypred[i] = s;
}

// ---------------- outputs ----------------
__global__ void __launch_bounds__(1024) k_loss(const float* __restrict__ ypred,
                                               const float* __restrict__ entsum,
                                               float* __restrict__ out) {
    __shared__ float s[1024];
    float a = 0.f;
    for (int i = threadIdx.x; i < NROW; i += 1024) {
        float d = ypred[i] - entsum[i] * (1.f / MAXIT);
        a += d * d;
    }
    s[threadIdx.x] = a;
    __syncthreads();
    for (int off = 512; off; off >>= 1) {
        if (threadIdx.x < off) s[threadIdx.x] += s[threadIdx.x + off];
        __syncthreads();
    }
    if (threadIdx.x == 0) out[0] = s[0] / NROW;
}

__global__ void k_mask(const int* __restrict__ mask, float* __restrict__ out) {
    int i = blockIdx.x * blockDim.x + threadIdx.x;
    if (i < NROW) out[1 + i] = mask[i] ? 1.f : 0.f;
}

extern "C" void kernel_launch(void* const* d_in, const int* in_sizes, int n_in,
                              void* d_out, int out_size, void* d_ws, size_t ws_size,
                              hipStream_t stream) {
    const float* emb   = (const float*)d_in[0];
    const int*   labels= (const int*)d_in[1];
    const int*   lmask = (const int*)d_in[2];
    const float* W[5]; const float* bvec[5]; const float* gam[5]; const float* bet[5];
    for (int l = 0; l < 5; ++l) {
        W[l]    = (const float*)d_in[3 + 4 * l];
        bvec[l] = (const float*)d_in[4 + 4 * l];
        gam[l]  = (const float*)d_in[5 + 4 * l];
        bet[l]  = (const float*)d_in[6 + 4 * l];
    }
    const float* Wf = (const float*)d_in[23];
    const float* bf = (const float*)d_in[24];
    (void)in_sizes; (void)n_in; (void)out_size; (void)ws_size;

    char* wsb = (char*)d_ws;
    size_t off = 0;
    auto carve = [&](size_t bytes) -> void* {
        void* p = wsb + off;
        off += (bytes + 255) & ~(size_t)255;
        return p;
    };
    __bf16* Amat  = (__bf16*)carve((size_t)NROW * NROW * 2);   // 128 MB (L2-resident)
    __bf16* embh  = (__bf16*)carve((size_t)NROW * EDIM * 2);   // 8 MB
    float*  sq    = (float*)carve(NROW * 4);
    float*  sigma = (float*)carve(NROW * 4);
    float*  Xa    = (float*)carve((size_t)NROW * NCLS * 4);
    float*  Xb    = (float*)carve((size_t)NROW * NCLS * 4);
    float*  entsum= (float*)carve(NROW * 4);
    int*    done  = (int*)carve(256);
    float*  errsq = (float*)carve(256);
    float*  actY  = (float*)carve((size_t)NROW * 256 * 4);
    float*  actF  = (float*)carve((size_t)NROW * 256 * 4);
    __bf16* actB  = (__bf16*)carve((size_t)NROW * 256 * 2);
    __bf16* wb    = (__bf16*)carve((size_t)256 * 512 * 2);
    float*  stats = (float*)carve(2 * 256 * 4);
    float*  ypred = (float*)carve(NROW * 4);

    // --- GTG target ---
    k_f32_to_bf16<<<(NROW * EDIM + 255) / 256, 256, 0, stream>>>(emb, embh, NROW * EDIM);
    k_rownorm<<<NROW / 8, 256, 0, stream>>>(emb, sq);
    k_sigma<<<NROW / 16, 128, 0, stream>>>(embh, sq, sigma);
    k_gram<<<dim3(NROW / 64, NROW / 64), 128, 0, stream>>>(embh, sq, sigma, Amat);
    k_init<<<(NROW + 255) / 256, 256, 0, stream>>>(labels, lmask, Xa, entsum, done, errsq);
    float* xo = Xa; float* xn = Xb;
    for (int it = 0; it < MAXIT; ++it) {
        k_replicator<<<NROW / 8, 256, 0, stream>>>(Amat, xo, xn, entsum, done, errsq);
        k_update<<<1, 1, 0, stream>>>(done, errsq);
        float* t = xo; xo = xn; xn = t;
    }

    // --- MLP ---
    const int din[5]  = {512, 256, 128, 64, 32};
    const int dout[5] = {256, 128, 64, 32, 16};
    const __bf16* cur = embh;
    for (int l = 0; l < 5; ++l) {
        const int K = din[l], O = dout[l];
        k_f32_to_bf16<<<(O * K + 255) / 256, 256, 0, stream>>>(W[l], wb, O * K);
        k_gemm_bias<<<dim3(O / 16, NROW / 64), 128, 0, stream>>>(cur, wb, bvec[l], actY, K, O);
        k_bn_stats<<<O, 256, 0, stream>>>(actY, stats, O);
        k_bn_apply<<<(NROW * O + 255) / 256, 256, 0, stream>>>(actY, stats, gam[l], bet[l],
                                                               actB, actF, O);
        cur = actB;
    }
    k_final<<<NROW / 256, 256, 0, stream>>>(actF, Wf, bf, ypred);

    // --- outputs: [loss, mask(8192)] ---
    k_loss<<<1, 1024, 0, stream>>>(ypred, entsum, (float*)d_out);
    k_mask<<<NROW / 256, 256, 0, stream>>>(lmask, (float*)d_out);
}
